// GATv4_40209483825152
// MI455X (gfx1250) — compile-verified, matrix operand order checked by reading, not run on hardware
//
#include <hip/hip_runtime.h>
#include <hip/hip_bf16.h>

// ---------------- problem constants (match reference) ----------------
#define NNODES   2000
#define BATCH    8
#define NTOT     16000          // B * NUM_NODES
#define INF_DIM  128
#define HC       128
#define NHEAD    4
#define CHAN     32
#define EDGES    512000
#define ETOT     528000         // EDGES + NTOT self loops
#define MSDIM    6000
#define FC1DIM   512
#define FC2DIM   128
#define NEGSLOPE 0.2f

typedef __attribute__((ext_vector_type(16))) _Float16 v16h;
typedef __attribute__((ext_vector_type(8)))  _Float16 v8h;
typedef __attribute__((ext_vector_type(8)))  float    v8f;

// ---------------- helpers ----------------
__device__ __forceinline__ float atomicMaxF(float* addr, float val) {
  // works when *addr initialized to -inf
  if (val >= 0.0f)
    return __int_as_float(atomicMax((int*)addr, __float_as_int(val)));
  else
    return __uint_as_float(atomicMin((unsigned int*)addr, __float_as_uint(val)));
}

// ---------------- fp32 -> fp16 conversion ----------------
__global__ void cvt_f16_kernel(const float* __restrict__ in,
                               _Float16* __restrict__ out, int n) {
  int i = blockIdx.x * blockDim.x + threadIdx.x;
  if (i < n) out[i] = (_Float16)in[i];
}

// ---------------- WMMA GEMM: out[M,128] = A[M,128] @ W[128,128]^T + bias ----
// One wave32 computes one 16x16 output tile. K = 128 = 4 WMMA steps.
// A-layout (16-bit 16x32): lane L holds row M=L%16; K halves selected by L>=16.
// B mirrors A with the output-channel row of W (since B[k][n] = W[n][k]).
__global__ void __launch_bounds__(256)
gemm_wmma_f16(const _Float16* __restrict__ A16,  // [M,128]
              const _Float16* __restrict__ W16,  // [128,128] (out_ch, k)
              const float*    __restrict__ bias, // [128]
              float*          __restrict__ outp, // [M,128]
              int M) {
  int wave = (blockIdx.x * blockDim.x + threadIdx.x) >> 5;
  int lane = threadIdx.x & 31;
  int mt = wave >> 3;          // M/16 tiles
  int nt = wave & 7;           // 8 N tiles
  if (mt * 16 >= M) return;

  int mrow = mt * 16 + (lane & 15);
  int ncol = nt * 16 + (lane & 15);
  int off  = (lane >> 4) << 3;           // 0 or 8 (K sub-half per lane group)

  const _Float16* arow = A16 + (size_t)mrow * HC;
  const _Float16* brow = W16 + (size_t)ncol * HC;

  float bv = bias[ncol];
  v8f c = {bv, bv, bv, bv, bv, bv, bv, bv};

#pragma unroll
  for (int kb = 0; kb < HC; kb += 32) {
    v8h alo = *(const v8h*)(arow + kb + off);
    v8h ahi = *(const v8h*)(arow + kb + 16 + off);
    v8h blo = *(const v8h*)(brow + kb + off);
    v8h bhi = *(const v8h*)(brow + kb + 16 + off);
    v16h a, b;
#pragma unroll
    for (int i = 0; i < 8; ++i) {
      a[i] = alo[i]; a[i + 8] = ahi[i];
      b[i] = blo[i]; b[i + 8] = bhi[i];
    }
    c = __builtin_amdgcn_wmma_f32_16x16x32_f16(
        /*neg_a=*/false, a, /*neg_b=*/false, b,
        /*c_mod=*/(short)0, c, /*reuse_a=*/false, /*reuse_b=*/false);
  }

  // C/D layout: VGPR v -> row v (lanes 0-15) / v+8 (lanes 16-31), col = lane%16
  int rbase = mt * 16 + ((lane >> 4) << 3);
#pragma unroll
  for (int v = 0; v < 8; ++v)
    outp[(size_t)(rbase + v) * HC + ncol] = c[v];
}

// ---------------- per-layer init: acc=0, amax=-inf, denom=0 ----------------
__global__ void init_layer_kernel(float* __restrict__ acc,
                                  float* __restrict__ amax,
                                  float* __restrict__ denom) {
  int i = blockIdx.x * blockDim.x + threadIdx.x;
  if (i < NTOT * HC) acc[i] = 0.0f;
  if (i < NTOT * NHEAD) { amax[i] = -__builtin_inff(); denom[i] = 0.0f; }
}

// ---------------- edge attention logits + segment max ----------------
// one wave per edge; lane covers 4 contiguous channels (h = lane/8)
__global__ void __launch_bounds__(256)
edge_attn_kernel(const float* __restrict__ xl, const float* __restrict__ xr,
                 const float* __restrict__ att, const int* __restrict__ esrc,
                 const int* __restrict__ edst, float* __restrict__ a_out,
                 float* __restrict__ amax) {
  int e = (blockIdx.x * blockDim.x + threadIdx.x) >> 5;
  int lane = threadIdx.x & 31;
  if (e >= ETOT) return;
  int s, d;
  if (e < EDGES) { s = esrc[e]; d = edst[e]; }
  else           { s = e - EDGES; d = s; }

  float4 xlv = *(const float4*)(xl + (size_t)s * HC + lane * 4);
  float4 xrv = *(const float4*)(xr + (size_t)d * HC + lane * 4);
  float4 av  = *(const float4*)(att + lane * 4);

  float p = 0.0f, t;
  t = xlv.x + xrv.x; t = t > 0.0f ? t : NEGSLOPE * t; p += av.x * t;
  t = xlv.y + xrv.y; t = t > 0.0f ? t : NEGSLOPE * t; p += av.y * t;
  t = xlv.z + xrv.z; t = t > 0.0f ? t : NEGSLOPE * t; p += av.z * t;
  t = xlv.w + xrv.w; t = t > 0.0f ? t : NEGSLOPE * t; p += av.w * t;

  p += __shfl_down(p, 4, 8);
  p += __shfl_down(p, 2, 8);
  p += __shfl_down(p, 1, 8);
  if ((lane & 7) == 0) {
    int h = lane >> 3;
    a_out[(size_t)e * NHEAD + h] = p;
    atomicMaxF(&amax[(size_t)d * NHEAD + h], p);
  }
}

// ---------------- exp(a - amax[dst]) and denominator ----------------
__global__ void edge_exp_kernel(float* __restrict__ a_ex,
                                const float* __restrict__ amax,
                                float* __restrict__ denom,
                                const int* __restrict__ edst) {
  int i = blockIdx.x * blockDim.x + threadIdx.x;
  if (i >= ETOT * NHEAD) return;
  int e = i >> 2, h = i & 3;
  int d = (e < EDGES) ? edst[e] : (e - EDGES);
  float ex = __expf(a_ex[i] - amax[(size_t)d * NHEAD + h]);
  a_ex[i] = ex;  // overwrite logits with exp in place
  atomicAdd(&denom[(size_t)d * NHEAD + h], ex);
}

// ---------------- alpha-weighted scatter of xl[src] into acc[dst] ----------
__global__ void edge_scatter_kernel(const float* __restrict__ ex,
                                    const float* __restrict__ denom,
                                    const float* __restrict__ xl,
                                    const int* __restrict__ esrc,
                                    const int* __restrict__ edst,
                                    float* __restrict__ acc) {
  long long i = (long long)blockIdx.x * blockDim.x + threadIdx.x;
  if (i >= (long long)ETOT * HC) return;
  int e = (int)(i >> 7);
  int t = (int)(i & 127);
  int h = t >> 5;
  int s, d;
  if (e < EDGES) { s = esrc[e]; d = edst[e]; }
  else           { s = e - EDGES; d = s; }
  float alpha = ex[(size_t)e * NHEAD + h] / denom[(size_t)d * NHEAD + h];
  atomicAdd(&acc[(size_t)d * HC + t], alpha * xl[(size_t)s * HC + t]);
}

// ---------------- bias + relu (in place) + optional f16 copy ----------------
__global__ void bias_relu_kernel(float* __restrict__ acc,
                                 const float* __restrict__ bias,
                                 _Float16* __restrict__ out16) {
  int i = blockIdx.x * blockDim.x + threadIdx.x;
  if (i >= NTOT * HC) return;
  float v = acc[i] + bias[i & (HC - 1)];
  v = v > 0.0f ? v : 0.0f;
  acc[i] = v;
  if (out16) out16[i] = (_Float16)v;
}

// ---------------- pooling: raw[layer*NTOT+n] = dot(h[n],W)+b or mean --------
__global__ void __launch_bounds__(256)
pool_kernel(const float* __restrict__ h, const float* __restrict__ W,
            const float* __restrict__ b, float* __restrict__ raw, int layer) {
  int n = (blockIdx.x * blockDim.x + threadIdx.x) >> 5;
  int lane = threadIdx.x & 31;
  if (n >= NTOT) return;
  float4 hv = *(const float4*)(h + (size_t)n * HC + lane * 4);
  float p;
  if (W) {
    float4 wv = *(const float4*)(W + lane * 4);
    p = hv.x * wv.x + hv.y * wv.y + hv.z * wv.z + hv.w * wv.w;
  } else {
    p = hv.x + hv.y + hv.z + hv.w;
  }
  p += __shfl_down(p, 16, 32);
  p += __shfl_down(p, 8, 32);
  p += __shfl_down(p, 4, 32);
  p += __shfl_down(p, 2, 32);
  p += __shfl_down(p, 1, 32);
  if (lane == 0)
    raw[(size_t)layer * NTOT + n] = W ? (p + b[0]) : (p * (1.0f / 128.0f));
}

// ---------------- LayerNorm over node dim (2000) per (b, layer) -------------
__global__ void __launch_bounds__(256)
ln_kernel(const float* __restrict__ raw, const float* __restrict__ g,
          const float* __restrict__ bb, float* __restrict__ ms) {
  int b = blockIdx.x & 7;
  int layer = blockIdx.x >> 3;
  const float* row = raw + (size_t)layer * NTOT + (size_t)b * NNODES;
  __shared__ float red[256];

  float s = 0.0f;
  for (int i = threadIdx.x; i < NNODES; i += 256) s += row[i];
  red[threadIdx.x] = s; __syncthreads();
  for (int o = 128; o >= 1; o >>= 1) {
    if (threadIdx.x < o) red[threadIdx.x] += red[threadIdx.x + o];
    __syncthreads();
  }
  float mean = red[0] * (1.0f / NNODES);
  __syncthreads();

  float v = 0.0f;
  for (int i = threadIdx.x; i < NNODES; i += 256) {
    float t = row[i] - mean; v += t * t;
  }
  red[threadIdx.x] = v; __syncthreads();
  for (int o = 128; o >= 1; o >>= 1) {
    if (threadIdx.x < o) red[threadIdx.x] += red[threadIdx.x + o];
    __syncthreads();
  }
  float rstd = rsqrtf(red[0] * (1.0f / NNODES) + 1e-5f);

  for (int i = threadIdx.x; i < NNODES; i += 256)
    ms[(size_t)b * MSDIM + (size_t)layer * NNODES + i] =
        (row[i] - mean) * rstd * g[i] + bb[i];
}

// ---------------- small FC: one output per block ----------------------------
__global__ void __launch_bounds__(256)
fc_kernel(const float* __restrict__ in, const float* __restrict__ W,
          const float* __restrict__ bias, float* __restrict__ out,
          int K, int O, int relu) {
  int o = blockIdx.x % O;
  int b = blockIdx.x / O;
  const float* x = in + (size_t)b * K;
  const float* w = W + (size_t)o * K;
  __shared__ float red[256];
  float s = 0.0f;
  for (int i = threadIdx.x; i < K; i += 256) s += x[i] * w[i];
  red[threadIdx.x] = s; __syncthreads();
  for (int off = 128; off >= 1; off >>= 1) {
    if (threadIdx.x < off) red[threadIdx.x] += red[threadIdx.x + off];
    __syncthreads();
  }
  if (threadIdx.x == 0) {
    float v = red[0] + bias[o];
    if (relu) v = v > 0.0f ? v : 0.0f;
    out[(size_t)b * O + o] = v;
  }
}

// ---------------- host launcher ----------------
extern "C" void kernel_launch(void* const* d_in, const int* in_sizes, int n_in,
                              void* d_out, int out_size, void* d_ws, size_t ws_size,
                              hipStream_t stream) {
  // inputs per setup_inputs() order
  const float* x    = (const float*)d_in[0];
  const int*   eidx = (const int*)d_in[1];
  const float* Wl1  = (const float*)d_in[2];  const float* bl1 = (const float*)d_in[3];
  const float* Wr1  = (const float*)d_in[4];  const float* br1 = (const float*)d_in[5];
  const float* att1 = (const float*)d_in[6];  const float* bias1 = (const float*)d_in[7];
  const float* Wl2  = (const float*)d_in[8];  const float* bl2 = (const float*)d_in[9];
  const float* Wr2  = (const float*)d_in[10]; const float* br2 = (const float*)d_in[11];
  const float* att2 = (const float*)d_in[12]; const float* bias2 = (const float*)d_in[13];
  const float* Wp1  = (const float*)d_in[14]; const float* bp1 = (const float*)d_in[15];
  const float* Wp2  = (const float*)d_in[16]; const float* bp2 = (const float*)d_in[17];
  const float* ln_g = (const float*)d_in[18]; const float* ln_b = (const float*)d_in[19];
  const float* We1  = (const float*)d_in[20]; const float* be1 = (const float*)d_in[21];
  const float* We2  = (const float*)d_in[22]; const float* be2 = (const float*)d_in[23];
  const float* We3  = (const float*)d_in[24]; const float* be3 = (const float*)d_in[25];

  const int* esrc = eidx;
  const int* edst = eidx + EDGES;

  // workspace carve-out
  char* base = (char*)d_ws;
  size_t off = 0;
  auto alloc = [&](size_t bytes) -> void* {
    void* p = base + off;
    off = (off + bytes + 255) & ~(size_t)255;
    return p;
  };
  _Float16* x16   = (_Float16*)alloc((size_t)NTOT * HC * 2);
  _Float16* h16   = (_Float16*)alloc((size_t)NTOT * HC * 2);
  _Float16* w16l1 = (_Float16*)alloc(HC * HC * 2);
  _Float16* w16r1 = (_Float16*)alloc(HC * HC * 2);
  _Float16* w16l2 = (_Float16*)alloc(HC * HC * 2);
  _Float16* w16r2 = (_Float16*)alloc(HC * HC * 2);
  float* xl    = (float*)alloc((size_t)NTOT * HC * 4);
  float* xr    = (float*)alloc((size_t)NTOT * HC * 4);
  float* acc   = (float*)alloc((size_t)NTOT * HC * 4);
  float* a_buf = (float*)alloc((size_t)ETOT * NHEAD * 4);
  float* amax  = (float*)alloc((size_t)NTOT * NHEAD * 4);
  float* denom = (float*)alloc((size_t)NTOT * NHEAD * 4);
  float* raw   = (float*)alloc((size_t)3 * NTOT * 4);
  float* ms    = (float*)alloc((size_t)BATCH * MSDIM * 4);
  float* z1    = (float*)alloc((size_t)BATCH * FC1DIM * 4);
  float* z2    = (float*)alloc((size_t)BATCH * FC2DIM * 4);
  (void)ws_size; (void)n_in; (void)in_sizes; (void)out_size;

  const int T = 256;
  const int nNodeElem = NTOT * HC;                 // 2,048,000
  const int gNodeElem = (nNodeElem + T - 1) / T;
  const int gGemm = NTOT / 16;                     // 1000 blocks, 8 waves each
  const int gEdgeWave = (ETOT * 32 + T - 1) / T;   // wave per edge
  const int gEdgeH = (ETOT * NHEAD + T - 1) / T;
  const long long scat = (long long)ETOT * HC;
  const int gScat = (int)((scat + T - 1) / T);
  const int gPool = (NTOT * 32 + T - 1) / T;

  // ---- precision conversion ----
  cvt_f16_kernel<<<gNodeElem, T, 0, stream>>>(x, x16, nNodeElem);
  cvt_f16_kernel<<<64, T, 0, stream>>>(Wl1, w16l1, HC * HC);
  cvt_f16_kernel<<<64, T, 0, stream>>>(Wr1, w16r1, HC * HC);
  cvt_f16_kernel<<<64, T, 0, stream>>>(Wl2, w16l2, HC * HC);
  cvt_f16_kernel<<<64, T, 0, stream>>>(Wr2, w16r2, HC * HC);

  // ---- x0 = mean(x, -1) ----
  pool_kernel<<<gPool, T, 0, stream>>>(x, nullptr, nullptr, raw, 0);

  // ================= layer 1 =================
  gemm_wmma_f16<<<gGemm, T, 0, stream>>>(x16, w16l1, bl1, xl, NTOT);
  gemm_wmma_f16<<<gGemm, T, 0, stream>>>(x16, w16r1, br1, xr, NTOT);
  init_layer_kernel<<<gNodeElem, T, 0, stream>>>(acc, amax, denom);
  edge_attn_kernel<<<gEdgeWave, T, 0, stream>>>(xl, xr, att1, esrc, edst, a_buf, amax);
  edge_exp_kernel<<<gEdgeH, T, 0, stream>>>(a_buf, amax, denom, edst);
  edge_scatter_kernel<<<gScat, T, 0, stream>>>(a_buf, denom, xl, esrc, edst, acc);
  bias_relu_kernel<<<gNodeElem, T, 0, stream>>>(acc, bias1, h16);
  pool_kernel<<<gPool, T, 0, stream>>>(acc, Wp1, bp1, raw, 1);

  // ================= layer 2 =================
  gemm_wmma_f16<<<gGemm, T, 0, stream>>>(h16, w16l2, bl2, xl, NTOT);
  gemm_wmma_f16<<<gGemm, T, 0, stream>>>(h16, w16r2, br2, xr, NTOT);
  init_layer_kernel<<<gNodeElem, T, 0, stream>>>(acc, amax, denom);
  edge_attn_kernel<<<gEdgeWave, T, 0, stream>>>(xl, xr, att2, esrc, edst, a_buf, amax);
  edge_exp_kernel<<<gEdgeH, T, 0, stream>>>(a_buf, amax, denom, edst);
  edge_scatter_kernel<<<gScat, T, 0, stream>>>(a_buf, denom, xl, esrc, edst, acc);
  bias_relu_kernel<<<gNodeElem, T, 0, stream>>>(acc, bias2, (_Float16*)nullptr);
  pool_kernel<<<gPool, T, 0, stream>>>(acc, Wp2, bp2, raw, 2);

  // ---- LayerNorm -> ms [B, 6000] ----
  ln_kernel<<<24, T, 0, stream>>>(raw, ln_g, ln_b, ms);

  // ---- MLP encoder ----
  fc_kernel<<<BATCH * FC1DIM, T, 0, stream>>>(ms, We1, be1, z1, MSDIM, FC1DIM, 1);
  fc_kernel<<<BATCH * FC2DIM, T, 0, stream>>>(z1, We2, be2, z2, FC1DIM, FC2DIM, 1);
  fc_kernel<<<BATCH, T, 0, stream>>>(z2, We3, be3, (float*)d_out, FC2DIM, 1, 0);
}
